// TestNet_81295140979494
// MI455X (gfx1250) — matrix operand order repaired and checked by reading, not compiled, not run on hardware
//
#include <hip/hip_runtime.h>
#include <stdint.h>

// Problem constants (from reference)
#define NB    4
#define NPTS  8192
#define NS    1024
#define NDIM  64

typedef _Float16 h8   __attribute__((ext_vector_type(8)));
typedef _Float16 v16h __attribute__((ext_vector_type(16)));
typedef float    v8f  __attribute__((ext_vector_type(8)));

// ---------------------------------------------------------------------------
// gfx1250 async global->LDS helpers (ASYNCcnt-tracked, per ISA 08_async_tensor)
// ---------------------------------------------------------------------------
__device__ __forceinline__ void async_g2l_b128(uint32_t lds_off, const void* gp) {
  asm volatile("global_load_async_to_lds_b128 %0, %1, off"
               :: "v"(lds_off), "v"((uint64_t)(uintptr_t)gp)
               : "memory");
}
__device__ __forceinline__ void wait_async0() {
  asm volatile("s_wait_asynccnt 0x0" ::: "memory");
}

// ---------------------------------------------------------------------------
// WMMA operand builders (CDNA5 ISA 7.12.2 layouts, wave32)
// A (16x32 f16): lanes 0-15: M=lane, K = kb*32 + {0..7, 16..23}
//                lanes 16-31: M=lane-16, K = kb*32 + {8..15, 24..31}
// ---------------------------------------------------------------------------
__device__ __forceinline__ v16h load_a16(const _Float16* __restrict__ p0,
                                         int pitch, int kb, int lane) {
  const int M = lane & 15;
  const int koff = (lane < 16) ? 0 : 8;
  const _Float16* p = p0 + M * pitch + kb * 32 + koff;
  h8 lo = *(const h8*)(p);        // K base + 0..7   (16B LDS load)
  h8 hi = *(const h8*)(p + 16);   // K base + 16..23 (16B LDS load)
  v16h a;
#pragma unroll
  for (int i = 0; i < 8; ++i) { a[i] = lo[i]; a[i + 8] = hi[i]; }
  return a;
}

// B (32x16 f16): lanes 0-15: N=lane, K = kb*32 + {0..7, 16..23}
//                lanes 16-31: N=lane-16, K = kb*32 + {8..15, 24..31}
__device__ __forceinline__ v16h load_b16(const _Float16* __restrict__ w,
                                         int pitch, int nb, int lane, int kb) {
  const int nn = lane & 15;
  const int koff = (lane < 16) ? 0 : 8;
  const _Float16* p = w + (nb * 16 + nn) * pitch + kb * 32 + koff;
  h8 lo = *(const h8*)(p);
  h8 hi = *(const h8*)(p + 16);
  v16h b;
#pragma unroll
  for (int i = 0; i < 8; ++i) { b[i] = lo[i]; b[i + 8] = hi[i]; }
  return b;
}

// One MLP layer on a 16-row tile: out = relu((in x W^T) * sc + sh)
// C/D layout: VGPR r -> M = r + (lane<16?0:8), N = lane&15.
template <int CPIN, int COUT>
__device__ __forceinline__ void mlp_layer(const _Float16* __restrict__ in, int ipitch,
                                          _Float16* __restrict__ out, int opitch,
                                          const _Float16* __restrict__ w,
                                          const float* __restrict__ sc,
                                          const float* __restrict__ sh, int lane) {
  const int mb = (lane < 16) ? 0 : 8;
  const int col0 = lane & 15;
#pragma unroll
  for (int nb = 0; nb < COUT / 16; ++nb) {
    v8f acc = {0.f, 0.f, 0.f, 0.f, 0.f, 0.f, 0.f, 0.f};
#pragma unroll
    for (int kb = 0; kb < CPIN / 32; ++kb) {
      v16h A  = load_a16(in, ipitch, kb, lane);
      v16h Bm = load_b16(w, CPIN, nb, lane, kb);
      acc = __builtin_amdgcn_wmma_f32_16x16x32_f16(false, A, false, Bm,
                                                   (short)0, acc, false, false);
    }
    const int col = nb * 16 + col0;
    const float scv = sc[col], shv = sh[col];
#pragma unroll
    for (int r2 = 0; r2 < 8; ++r2) {
      float v = fmaxf(acc[r2] * scv + shv, 0.f);
      out[(mb + r2) * opitch + col] = (_Float16)v;
    }
  }
}

// ---------------------------------------------------------------------------
// Weight packing: f32 [Cout][Cin] -> f16 [Cout][CP] zero-padded
// ---------------------------------------------------------------------------
__global__ void pack_kernel(const float* __restrict__ W, _Float16* __restrict__ dst,
                            int cout, int cin, int cp) {
  int i = blockIdx.x * blockDim.x + threadIdx.x;
  if (i >= cout * cp) return;
  int o = i / cp, c = i % cp;
  dst[i] = (c < cin) ? (_Float16)W[o * cin + c] : (_Float16)0.f;
}

// ---------------------------------------------------------------------------
// points (B,D,N) f32 -> ptsT (B,N,D) f16 ; reads coalesced along N.
// ---------------------------------------------------------------------------
__global__ void __launch_bounds__(256) transpose_pts_kernel(
    const float* __restrict__ pts, _Float16* __restrict__ ptsT) {
  const int t = blockIdx.x * blockDim.x + threadIdx.x;   // over B*N
  const int b = t >> 13, n = t & (NPTS - 1);
  const float* src = pts + (size_t)b * NDIM * NPTS + n;
  _Float16* dst = ptsT + (size_t)t * NDIM;
#pragma unroll
  for (int c8 = 0; c8 < NDIM / 8; ++c8) {
    h8 v;
#pragma unroll
    for (int j = 0; j < 8; ++j) v[j] = (_Float16)src[(size_t)(c8 * 8 + j) * NPTS];
    *(h8*)(dst + c8 * 8) = v;
  }
}

// ---------------------------------------------------------------------------
// FPS: one 1024-thread workgroup per batch; xyz fully LDS-resident.
// ---------------------------------------------------------------------------
__global__ void __launch_bounds__(1024) fps_kernel(const float* __restrict__ xyz,
                                                   float* __restrict__ newxyz,
                                                   float* __restrict__ out0) {
  extern __shared__ float fsm[];
  float* lx = fsm;
  float* ly = fsm + NPTS;
  float* lz = fsm + 2 * NPTS;
  float* redd = fsm + 3 * NPTS;                 // 32
  int*   redi = (int*)(fsm + 3 * NPTS + 32);    // 32
  int*   fpsl = (int*)(fsm + 3 * NPTS + 64);    // NS
  int*   bc   = fpsl + NS;                      // 1

  const int b = blockIdx.x, t = threadIdx.x;
  const float* xb = xyz + (size_t)b * 3 * NPTS;
  for (int i = t; i < NPTS; i += 1024) {
    lx[i] = xb[i];
    ly[i] = xb[NPTS + i];
    lz[i] = xb[2 * NPTS + i];
  }
  __syncthreads();

  float dist[8];
#pragma unroll
  for (int j = 0; j < 8; ++j) dist[j] = 1e10f;
  int far = 0;
  const int lane = t & 31, w = t >> 5;

  for (int it = 0; it < NS; ++it) {
    if (t == 0) fpsl[it] = far;
    const float cx = lx[far], cy = ly[far], cz = lz[far];
    float bestd = -1.f;
    int besti = 0;
#pragma unroll
    for (int j = 0; j < 8; ++j) {
      const int n = t + j * 1024;
      float dx = lx[n] - cx, dy = ly[n] - cy, dz = lz[n] - cz;
      float d = dx * dx + dy * dy + dz * dz;
      dist[j] = fminf(dist[j], d);
      if (dist[j] > bestd) { bestd = dist[j]; besti = n; }
    }
    // wave32 argmax (first-index tie-break, matching jnp.argmax)
    for (int off = 16; off; off >>= 1) {
      float od = __shfl_down(bestd, off, 32);
      int   oi = __shfl_down(besti, off, 32);
      if (od > bestd || (od == bestd && oi < besti)) { bestd = od; besti = oi; }
    }
    if (lane == 0) { redd[w] = bestd; redi[w] = besti; }
    __syncthreads();
    if (t < 32) {
      bestd = redd[t]; besti = redi[t];
      for (int off = 16; off; off >>= 1) {
        float od = __shfl_down(bestd, off, 32);
        int   oi = __shfl_down(besti, off, 32);
        if (od > bestd || (od == bestd && oi < besti)) { bestd = od; besti = oi; }
      }
      if (t == 0) *bc = besti;
    }
    __syncthreads();
    far = *bc;
  }
  __syncthreads();

  const int s = t;
  const int i = fpsl[s];
  const float px = lx[i], py = ly[i], pz = lz[i];
  float* nx = newxyz + ((size_t)b * NS + s) * 3;
  nx[0] = px; nx[1] = py; nx[2] = pz;
  out0[((size_t)b * 3 + 0) * NS + s] = px;
  out0[((size_t)b * 3 + 1) * NS + s] = py;
  out0[((size_t)b * 3 + 2) * NS + s] = pz;
}

// ---------------------------------------------------------------------------
// Ball query: one wave per query, 3 radii at once via wave32 ballot.
// ---------------------------------------------------------------------------
__global__ void __launch_bounds__(256) ball_kernel(const float* __restrict__ xyz,
                                                   const float* __restrict__ newxyz,
                                                   int* __restrict__ idx0,
                                                   int* __restrict__ idx1,
                                                   int* __restrict__ idx2) {
  const int wave = threadIdx.x >> 5, lane = threadIdx.x & 31;
  const int q = blockIdx.x * 8 + wave;
  const int b = q >> 10;
  const float qx = newxyz[q * 3 + 0];
  const float qy = newxyz[q * 3 + 1];
  const float qz = newxyz[q * 3 + 2];
  const float* xb = xyz + (size_t)b * 3 * NPTS;
  int* o0 = idx0 + (size_t)q * 16;
  int* o1 = idx1 + (size_t)q * 32;
  int* o2 = idx2 + (size_t)q * 64;

  int c0 = 0, c1 = 0, c2 = 0;
  int f0 = 0, f1 = 0, f2 = 0;
  bool g0 = false, g1 = false, g2 = false;
  const unsigned below = (1u << lane) - 1u;

  for (int base = 0; base < NPTS; base += 32) {
    const int n = base + lane;
    float dx = xb[n] - qx, dy = xb[NPTS + n] - qy, dz = xb[2 * NPTS + n] - qz;
    float d = dx * dx + dy * dy + dz * dz;
    unsigned m0 = (unsigned)__ballot(d <= 0.01f);   // r=0.1
    unsigned m1 = (unsigned)__ballot(d <= 0.04f);   // r=0.2
    unsigned m2 = (unsigned)__ballot(d <= 0.16f);   // r=0.4
    if (!g0 && m0) { f0 = base + __ffs((int)m0) - 1; g0 = true; }
    if (!g1 && m1) { f1 = base + __ffs((int)m1) - 1; g1 = true; }
    if (!g2 && m2) { f2 = base + __ffs((int)m2) - 1; g2 = true; }
    if (c0 < 16 && ((m0 >> lane) & 1u)) { int o = c0 + __popc(m0 & below); if (o < 16) o0[o] = n; }
    if (c1 < 32 && ((m1 >> lane) & 1u)) { int o = c1 + __popc(m1 & below); if (o < 32) o1[o] = n; }
    if (c2 < 64 && ((m2 >> lane) & 1u)) { int o = c2 + __popc(m2 & below); if (o < 64) o2[o] = n; }
    c0 = min(16, c0 + __popc(m0));
    c1 = min(32, c1 + __popc(m1));
    c2 = min(64, c2 + __popc(m2));
    if (c0 >= 16 && c1 >= 32 && c2 >= 64) break;
  }
  for (int i = c0 + lane; i < 16; i += 32) o0[i] = f0;
  for (int i = c1 + lane; i < 32; i += 32) o1[i] = f1;
  for (int i = c2 + lane; i < 64; i += 32) o2[i] = f2;
}

// ---------------------------------------------------------------------------
// Branch kernel: one wave per query point.
//   async-gather (f16 rows) -> L1 -> L2 -> L3(WMMA, regs) -> max over K.
// Per-wave LDS: bufG (16x96 gather, zero pad written once),
//               bufB (16x64 L1 out), bufC (16x96 L2 out).
// ---------------------------------------------------------------------------
template <int KS, int W1, int W2, int W3, int CHOFF>
__global__ void __launch_bounds__(256) branch_kernel(
    const float* __restrict__ xyz, const _Float16* __restrict__ ptsT,
    const float* __restrict__ newxyz, const int* __restrict__ idxb,
    const _Float16* __restrict__ wpack,
    const float* __restrict__ sc1, const float* __restrict__ sh1,
    const float* __restrict__ sc2, const float* __restrict__ sh2,
    const float* __restrict__ sc3, const float* __restrict__ sh3,
    float* __restrict__ out1) {
  extern __shared__ _Float16 smem[];
  constexpr int WTOT = W1 * 96 + W2 * W1 + W3 * W2;   // halves (16B multiple)
  _Float16* wl1 = smem;                 // [W1][96]
  _Float16* wl2 = wl1 + W1 * 96;        // [W2][W1]
  _Float16* wl3 = wl2 + W2 * W1;        // [W3][W2]
  _Float16* fb  = wl3 + WTOT - (W1 * 96 + W2 * W1) + (W1 * 96 + W2 * W1); // == wl3+WTOT (reset below)
  fb = smem + WTOT;

  const int tid = threadIdx.x;
  const int wave = tid >> 5, lane = tid & 31;
  _Float16* bufG = fb + wave * (16 * 96 + 16 * 64 + 16 * 96);
  _Float16* bufB = bufG + 16 * 96;
  _Float16* bufC = bufB + 16 * 64;

  const uint32_t ldsdyn = __builtin_amdgcn_groupstaticsize();

  // ---- stage all branch weights LDS-resident via async DMA ----
  {
    const char* src = (const char*)wpack;
    for (int i = tid; i < WTOT / 8; i += 256)   // 16B chunks
      async_g2l_b128(ldsdyn + (uint32_t)i * 16u, src + (size_t)i * 16);
    wait_async0();            // each wave drains its own ASYNCcnt
  }
  __syncthreads();

  const int q = blockIdx.x * 8 + wave;
  const int b = q >> 10, s = q & 1023;
  const float qx = newxyz[q * 3 + 0];
  const float qy = newxyz[q * 3 + 1];
  const float qz = newxyz[q * 3 + 2];

  const int r = lane & 15, p = lane >> 4;

  // zero pad region (cols 64..95) once; cols 64..66 rewritten per tile
  {
    uint32_t* z = (uint32_t*)(bufG + r * 96 + 64 + p * 16);
#pragma unroll
    for (int j = 0; j < 8; ++j) z[j] = 0u;
  }

  float vmax[(W3 / 16) * 8];
#pragma unroll
  for (int i = 0; i < (W3 / 16) * 8; ++i) vmax[i] = 0.f;  // relu outputs >= 0

  const uint32_t goff = ldsdyn +
      (uint32_t)((const char*)(bufG + r * 96 + p * 32) - (const char*)smem);

  for (int rt = 0; rt < KS / 16; ++rt) {
    // async-gather 16 samples x 64 f16 channels straight into LDS
    const int n = idxb[(size_t)q * KS + rt * 16 + r];
    const _Float16* srow = ptsT + ((size_t)b * NPTS + n) * NDIM + p * 32;
#pragma unroll
    for (int j = 0; j < 4; ++j)
      async_g2l_b128(goff + (uint32_t)j * 16u, srow + j * 8);
    if (p == 0) {   // relative xyz, cols 64..66
      float vx = xyz[((size_t)b * 3 + 0) * NPTS + n] - qx;
      float vy = xyz[((size_t)b * 3 + 1) * NPTS + n] - qy;
      float vz = xyz[((size_t)b * 3 + 2) * NPTS + n] - qz;
      bufG[r * 96 + 64] = (_Float16)vx;
      bufG[r * 96 + 65] = (_Float16)vy;
      bufG[r * 96 + 66] = (_Float16)vz;
    }
    wait_async0();

    mlp_layer<96, W1>(bufG, 96, bufB, W1, wl1, sc1, sh1, lane);
    mlp_layer<W1, W2>(bufB, W1, bufC, W2, wl2, sc2, sh2, lane);

    // final layer in registers, fused with running max over K
    const int col0 = lane & 15;
#pragma unroll
    for (int nb = 0; nb < W3 / 16; ++nb) {
      v8f acc = {0.f, 0.f, 0.f, 0.f, 0.f, 0.f, 0.f, 0.f};
#pragma unroll
      for (int kb = 0; kb < W2 / 32; ++kb) {
        v16h A  = load_a16(bufC, W2, kb, lane);
        v16h Bm = load_b16(wl3, W2, nb, lane, kb);
        acc = __builtin_amdgcn_wmma_f32_16x16x32_f16(false, A, false, Bm,
                                                     (short)0, acc, false, false);
      }
      const int col = nb * 16 + col0;
      const float scv = sc3[col], shv = sh3[col];
#pragma unroll
      for (int r2 = 0; r2 < 8; ++r2) {
        float v = fmaxf(acc[r2] * scv + shv, 0.f);
        vmax[nb * 8 + r2] = fmaxf(vmax[nb * 8 + r2], v);
      }
    }
  }

  // reduce 16 rows (M in {0..7} here, {8..15} in lane^16) and write
#pragma unroll
  for (int nb = 0; nb < W3 / 16; ++nb) {
    float m = vmax[nb * 8];
#pragma unroll
    for (int r2 = 1; r2 < 8; ++r2) m = fmaxf(m, vmax[nb * 8 + r2]);
    m = fmaxf(m, __shfl_xor(m, 16, 32));
    if (lane < 16) {
      const int ch = CHOFF + nb * 16 + lane;
      out1[((size_t)b * 320 + ch) * NS + s] = m;
    }
  }
}

// ---------------------------------------------------------------------------
// Host-side orchestration
// ---------------------------------------------------------------------------
extern "C" void kernel_launch(void* const* d_in, const int* in_sizes, int n_in,
                              void* d_out, int out_size, void* d_ws, size_t ws_size,
                              hipStream_t stream) {
  (void)in_sizes; (void)n_in; (void)out_size; (void)ws_size;
  const float* xyz = (const float*)d_in[0];
  const float* pts = (const float*)d_in[1];
  float* out0 = (float*)d_out;                 // (B,3,NS)
  float* out1 = out0 + (size_t)NB * 3 * NS;    // (B,320,NS)

  char* ws = (char*)d_ws;
  size_t off = 0;
  float* newxyz = (float*)(ws + off); off += (size_t)NB * NS * 3 * sizeof(float);
  int* idx0 = (int*)(ws + off); off += (size_t)NB * NS * 16 * sizeof(int);
  int* idx1 = (int*)(ws + off); off += (size_t)NB * NS * 32 * sizeof(int);
  int* idx2 = (int*)(ws + off); off += (size_t)NB * NS * 64 * sizeof(int);
  off = (off + 255) & ~(size_t)255;
  _Float16* ptsT = (_Float16*)(ws + off); off += (size_t)NB * NPTS * NDIM * sizeof(_Float16);
  off = (off + 255) & ~(size_t)255;
  _Float16* wpack = (_Float16*)(ws + off);

  static const int WW[3][3]  = {{32, 32, 64}, {64, 64, 128}, {64, 96, 128}};
  static const int CIN[3][3] = {{67, 32, 32}, {67, 64, 64}, {67, 64, 96}};
  static const int CP[3][3]  = {{96, 32, 32}, {96, 64, 64}, {96, 64, 96}};

  _Float16* wbase[3];
  size_t hoff = 0;
  for (int br = 0; br < 3; ++br) {
    wbase[br] = wpack + hoff;
    for (int ly = 0; ly < 3; ++ly) {
      const float* W = (const float*)d_in[2 + br * 9 + ly * 3 + 0];
      const int cout = WW[br][ly], cin = CIN[br][ly], cp = CP[br][ly];
      const int nel = cout * cp;
      pack_kernel<<<(nel + 255) / 256, 256, 0, stream>>>(W, wpack + hoff, cout, cin, cp);
      hoff += (size_t)nel;
    }
  }

  transpose_pts_kernel<<<(NB * NPTS) / 256, 256, 0, stream>>>(pts, ptsT);

  const size_t fps_sh = (size_t)(3 * NPTS) * 4 + 64 * 4 + (size_t)NS * 4 + 16;
  fps_kernel<<<NB, 1024, fps_sh, stream>>>(xyz, newxyz, out0);

  ball_kernel<<<(NB * NS) / 8, 256, 0, stream>>>(xyz, newxyz, idx0, idx1, idx2);

  const float* sc[3][3], *sh[3][3];
  for (int br = 0; br < 3; ++br)
    for (int ly = 0; ly < 3; ++ly) {
      sc[br][ly] = (const float*)d_in[2 + br * 9 + ly * 3 + 1];
      sh[br][ly] = (const float*)d_in[2 + br * 9 + ly * 3 + 2];
    }

  auto shbytes = [](int w1, int w2, int w3) -> size_t {
    return (size_t)((w1 * 96 + w2 * w1 + w3 * w2) +
                    8 * (16 * 96 + 16 * 64 + 16 * 96)) * sizeof(_Float16);
  };

  branch_kernel<16, 32, 32, 64, 0><<<(NB * NS) / 8, 256, shbytes(32, 32, 64), stream>>>(
      xyz, ptsT, newxyz, idx0, wbase[0],
      sc[0][0], sh[0][0], sc[0][1], sh[0][1], sc[0][2], sh[0][2], out1);
  branch_kernel<32, 64, 64, 128, 64><<<(NB * NS) / 8, 256, shbytes(64, 64, 128), stream>>>(
      xyz, ptsT, newxyz, idx1, wbase[1],
      sc[1][0], sh[1][0], sc[1][1], sh[1][1], sc[1][2], sh[1][2], out1);
  branch_kernel<64, 64, 96, 128, 192><<<(NB * NS) / 8, 256, shbytes(64, 96, 128), stream>>>(
      xyz, ptsT, newxyz, idx2, wbase[2],
      sc[2][0], sh[2][0], sc[2][1], sh[2][1], sc[2][2], sh[2][2], out1);
}
